// PatchMamba_36919538876641
// MI455X (gfx1250) — matrix operand order repaired and checked by reading, not compiled, not run on hardware
//
#include <hip/hip_runtime.h>
#include <hip/hip_bf16.h>
#include <math.h>

// ---------------------------------------------------------------------------
// PatchMamba forward for gfx1250 (MI455X).  All dense projections go through
// v_wmma_f32_16x16x32_f16 (f32 data converted to f16 on load).  Row-major
// GEMMs compute a 16x64 strip per wave (4 WMMA accumulators) so the A
// fragment is loaded once per 4 tiles; all dims are clean (M%16, N%64, K%32)
// via zero-padded weight staging, so operand loads are unguarded b128.
// Selective scan: one lane per (channel,state), shfl reduction, prefetch.
// ---------------------------------------------------------------------------

typedef __attribute__((ext_vector_type(16))) _Float16 v16h;
typedef __attribute__((ext_vector_type(8)))  float    v8f;

#define D_MODEL 128
#define D_INNER 256
#define D_STATE 16
#define DT_RANK 8
#define KDIR    4
#define NLAYER  2
#define HB      64          // block height/width after 2x2 patching
#define LSEQ    4096        // HB*HB
#define NTOK    8192        // B(2) * LSEQ
#define RPAD    48          // x_dbl rows padded (40 -> 48)
#define DTKP    32          // dt_proj K padded (8 -> 32)

__device__ __forceinline__ float siluf(float x)  { return x / (1.f + __expf(-x)); }
__device__ __forceinline__ float geluf(float x)  { return 0.5f * x * (1.f + erff(x * 0.70710678118f)); }
// stable softplus on the fast transcendental path (v_exp_f32 / v_log_f32)
__device__ __forceinline__ float softpf(float x)
{
    return fmaxf(x, 0.f) + __logf(1.f + __expf(-fabsf(x)));
}

__device__ __forceinline__ void cvt8(v16h& v, int base, float4 p, float4 q)
{
    v[base + 0] = (_Float16)p.x; v[base + 1] = (_Float16)p.y;
    v[base + 2] = (_Float16)p.z; v[base + 3] = (_Float16)p.w;
    v[base + 4] = (_Float16)q.x; v[base + 5] = (_Float16)q.y;
    v[base + 6] = (_Float16)q.z; v[base + 7] = (_Float16)q.w;
}

__device__ __forceinline__ float actf(float v, int act)
{
    if (act == 1) return geluf(v);
    if (act == 2) return softpf(v);
    return v;
}

// ---------------------------------------------------------------------------
// Row-major WMMA GEMM: C[m,n] = act(sum_k A[m,k]*B[n,k] + bias[n]) + Cadd[m,n]
// One wave per block computes a 16x64 strip (4 tiles, A loaded once/4 tiles).
// Requires M%16==0, N%64==0, Kdim%32==0.  act: 0=none, 1=gelu, 2=softplus
// ---------------------------------------------------------------------------
__global__ void gemm_wmma_rm(const float* __restrict__ A, int lda,
                             const float* __restrict__ Bm, int ldb,
                             const float* __restrict__ bias,
                             const float* __restrict__ Cadd, int ldr,
                             float* __restrict__ Cmat, int ldc,
                             int Kdim, int act)
{
    const int lane = threadIdx.x & 31;
    const int m0 = blockIdx.x * 16;
    const int n0 = blockIdx.y * 64;

    const int r  = lane & 15;
    const int hi = lane >> 4;
    const int kb = hi << 3;                       // half of A kbase (A kbase = 2*kb: 0 or 16)
    const float* Arow = A + (size_t)(m0 + r) * lda;
    const float* Brow0 = Bm + (size_t)(n0 + r) * ldb;

    v8f acc[4] = {v8f{}, v8f{}, v8f{}, v8f{}};
    for (int k0 = 0; k0 < Kdim; k0 += 32) {
        // A 16x32 f16 layout: e -> K = k0 + 16*hi + e  (16 contiguous)
        const float4* ap0 = (const float4*)(Arow + k0 + 2 * kb);
        const float4* ap1 = (const float4*)(Arow + k0 + 2 * kb + 8);
        float4 a0 = ap0[0], a1 = ap0[1], a2 = ap1[0], a3 = ap1[1];
        v16h av;
        cvt8(av, 0, a0, a1); cvt8(av, 8, a2, a3);
#pragma unroll
        for (int t = 0; t < 4; ++t) {
            // B 32x16 f16 layout: e -> K = k0 + 16*hi + e  (16 contiguous)
            const float4* bp = (const float4*)(Brow0 + (size_t)(t * 16) * ldb
                                               + k0 + (hi << 4));
            float4 b0 = bp[0], b1 = bp[1], b2 = bp[2], b3 = bp[3];
            v16h bv;
            cvt8(bv, 0, b0, b1); cvt8(bv, 8, b2, b3);
            acc[t] = __builtin_amdgcn_wmma_f32_16x16x32_f16(false, av, false, bv,
                                                            (short)0, acc[t],
                                                            false, false);
        }
    }

    // D layout: VGPR i, lanes 0-15 -> M=m0+i ; lanes 16-31 -> M=m0+8+i
    const int mbase = m0 + (hi << 3);
#pragma unroll
    for (int t = 0; t < 4; ++t) {
        const int ncol = n0 + t * 16 + r;
        const float bb = bias ? bias[ncol] : 0.f;
#pragma unroll
        for (int i = 0; i < 8; ++i) {
            int m = mbase + i;
            float v = actf(acc[t][i] + bb, act);
            if (Cadd) v += Cadd[(size_t)m * ldr + ncol];
            Cmat[(size_t)m * ldc + ncol] = v;
        }
    }
}

// ---------------------------------------------------------------------------
// Strided-A WMMA GEMM for [D,L]-major activations:
//   A element (m,k) at A[m + k*as_k]   (m contiguous -> lane-coalesced)
//   C element (m,n) at C[m + n*cs_n]
// B row-major [N,ldb].  M%16==0, N%16==0 (per wave), Kdim%32==0.
// ---------------------------------------------------------------------------
__global__ void gemm_wmma_sk(const float* __restrict__ A, long as_k,
                             const float* __restrict__ Bm, int ldb,
                             const float* __restrict__ bias,
                             float* __restrict__ Cmat, long cs_n,
                             int N, int Kdim, int act)
{
    const int lane = threadIdx.x & 31;
    const int wave = threadIdx.x >> 5;
    const int m0 = blockIdx.x * 16;
    const int n0 = (blockIdx.y * 4 + wave) * 16;
    if (n0 >= N) return;                          // wave-uniform: EXEC all-1s

    const int r  = lane & 15;
    const int hi = lane >> 4;
    const int kb = hi << 3;
    const float* Am   = A  + (m0 + r);
    const float* Brow = Bm + (size_t)(n0 + r) * ldb;

    v8f acc = {};
    for (int k0 = 0; k0 < Kdim; k0 += 32) {
        v16h av, bv;
#pragma unroll
        for (int e = 0; e < 8; ++e) {
            av[e]     = (_Float16)Am[(size_t)(k0 + 2 * kb + e) * as_k];
            av[e + 8] = (_Float16)Am[(size_t)(k0 + 2 * kb + 8 + e) * as_k];
        }
        const float4* bp = (const float4*)(Brow + k0 + (hi << 4));
        float4 b0 = bp[0], b1 = bp[1], b2 = bp[2], b3 = bp[3];
        cvt8(bv, 0, b0, b1); cvt8(bv, 8, b2, b3);
        acc = __builtin_amdgcn_wmma_f32_16x16x32_f16(false, av, false, bv,
                                                     (short)0, acc, false, false);
    }

    const int ncol  = n0 + r;
    const int mbase = m0 + (hi << 3);
    const float bb = bias ? bias[ncol] : 0.f;
#pragma unroll
    for (int i = 0; i < 8; ++i) {
        float v = actf(acc[i] + bb, act);
        Cmat[(size_t)(mbase + i) + (size_t)ncol * cs_n] = v;
    }
}

// Zero-pad weight staging --------------------------------------------------
__global__ void pad_rows(const float* __restrict__ src, float* __restrict__ dst,
                         int batch, int rows0, int rowsP, int cols)
{
    long idx = (long)blockIdx.x * blockDim.x + threadIdx.x;
    long total = (long)batch * rowsP * cols;
    if (idx >= total) return;
    int c = idx % cols; long t = idx / cols;
    int rp = t % rowsP; int b = t / rowsP;
    dst[idx] = (rp < rows0) ? src[((size_t)b * rows0 + rp) * cols + c] : 0.f;
}

__global__ void pad_cols(const float* __restrict__ src, float* __restrict__ dst,
                         long rows, int cols0, int colsP)
{
    long idx = (long)blockIdx.x * blockDim.x + threadIdx.x;
    if (idx >= rows * colsP) return;
    int cp = idx % colsP; long rr = idx / colsP;
    dst[idx] = (cp < cols0) ? src[rr * cols0 + cp] : 0.f;
}

// ---------------------------------------------------------------------------
// Direct 3x3 conv, pad=1 (init conv 6->32 and final conv 32->3)
// ---------------------------------------------------------------------------
__global__ void conv3x3(const float* __restrict__ in, const float* __restrict__ w,
                        const float* __restrict__ bias, float* __restrict__ out,
                        int B, int Cin, int Cout, int H, int W)
{
    long idx = (long)blockIdx.x * blockDim.x + threadIdx.x;
    long total = (long)B * Cout * H * W;
    if (idx >= total) return;
    int x = idx % W; long t = idx / W;
    int y = t % H;  t /= H;
    int co = t % Cout; int b = t / Cout;
    float s = bias ? bias[co] : 0.f;
    for (int ci = 0; ci < Cin; ++ci) {
        const float* ip = in + (((long)b * Cin + ci) * H) * W;
        const float* wp = w + ((long)co * Cin + ci) * 9;
        for (int dy = -1; dy <= 1; ++dy) {
            int yy = y + dy; if (yy < 0 || yy >= H) continue;
            for (int dx = -1; dx <= 1; ++dx) {
                int xx = x + dx; if (xx < 0 || xx >= W) continue;
                s += ip[(long)yy * W + xx] * wp[(dy + 1) * 3 + (dx + 1)];
            }
        }
    }
    out[idx] = s;
}

// x32[B,32,128,128] -> tok[NTOK,128]  (2x2 patch-to-channel)
__global__ void block_to_tokens(const float* __restrict__ x32, float* __restrict__ tok)
{
    long idx = (long)blockIdx.x * blockDim.x + threadIdx.x;
    if (idx >= (long)NTOK * 128) return;
    int c128 = idx % 128; long m = idx / 128;
    int j = m % HB; int i = (m / HB) % HB; int b = m / LSEQ;
    int ch = c128 & 31, blk = c128 >> 5, rr = blk >> 1, cc = blk & 1;
    tok[idx] = x32[(((size_t)b * 32 + ch) * 128 + rr * HB + i) * 128 + cc * HB + j];
}

// hc[NTOK,128] -> m32[B,32,128,128]  (inverse patching)
__global__ void tokens_to_image(const float* __restrict__ tok, float* __restrict__ m32)
{
    long idx = (long)blockIdx.x * blockDim.x + threadIdx.x;
    if (idx >= (long)NTOK * 128) return;
    int x = idx % 128; long t = idx / 128;
    int y = t % 128; t /= 128;
    int ch = t % 32; int b = t / 32;
    int rr = y / HB, i = y % HB, cc = x / HB, j = x % HB;
    m32[idx] = tok[((size_t)b * LSEQ + (size_t)i * HB + j) * 128 + ((rr * 2 + cc) * 32 + ch)];
}

// LayerNorm over C (one wave per token), optional SiLU(z) gate with row stride.
__global__ void layernorm_k(const float* __restrict__ in,
                            const float* __restrict__ gamma, const float* __restrict__ beta,
                            const float* __restrict__ gate, long gstride,
                            float* __restrict__ out, int M, int C)
{
    int tok = blockIdx.x * (blockDim.x >> 5) + (threadIdx.x >> 5);
    if (tok >= M) return;
    int lane = threadIdx.x & 31;
    const float* row = in + (size_t)tok * C;
    float s = 0.f, s2 = 0.f;
    for (int c = lane; c < C; c += 32) { float v = row[c]; s += v; s2 += v * v; }
    for (int m = 16; m >= 1; m >>= 1) { s += __shfl_xor(s, m); s2 += __shfl_xor(s2, m); }
    float mu = s / C, var = s2 / C - mu * mu;
    float inv = rsqrtf(var + 1e-5f);
    float* orow = out + (size_t)tok * C;
    for (int c = lane; c < C; c += 32) {
        float v = (row[c] - mu) * inv * gamma[c] + beta[c];
        if (gate) v *= siluf(gate[(size_t)tok * gstride + c]);
        orow[c] = v;
    }
}

// xz[NTOK,512] first half -> xin image [B,256,64,64]
__global__ void xz_to_image(const float* __restrict__ xz, float* __restrict__ img)
{
    long idx = (long)blockIdx.x * blockDim.x + threadIdx.x;
    if (idx >= (long)2 * D_INNER * LSEQ) return;
    int j = idx % HB; long t = idx / HB;
    int i = t % HB; t /= HB;
    int d = t % D_INNER; int b = t / D_INNER;
    img[idx] = xz[((size_t)b * LSEQ + (size_t)i * HB + j) * 512 + d];
}

// depthwise 3x3 + bias + SiLU on [B,256,64,64]; scatter into 4 scan directions
__global__ void dwconv_silu_xs(const float* __restrict__ img, const float* __restrict__ w,
                               const float* __restrict__ bias, float* __restrict__ xs)
{
    long idx = (long)blockIdx.x * blockDim.x + threadIdx.x;
    if (idx >= (long)2 * D_INNER * LSEQ) return;
    int j = idx % HB; long t = idx / HB;
    int i = t % HB; t /= HB;
    int d = t % D_INNER; int b = t / D_INNER;
    const float* ip = img + (((size_t)b * D_INNER + d) * HB) * HB;
    const float* wp = w + (size_t)d * 9;
    float s = bias[d];
    for (int dy = -1; dy <= 1; ++dy) {
        int yy = i + dy; if (yy < 0 || yy >= HB) continue;
        for (int dx = -1; dx <= 1; ++dx) {
            int xx = j + dx; if (xx < 0 || xx >= HB) continue;
            s += ip[(long)yy * HB + xx] * wp[(dy + 1) * 3 + (dx + 1)];
        }
    }
    float v = siluf(s);
    int l = i * HB + j, lt = j * HB + i;
    float* base = xs + ((size_t)b * KDIR) * D_INNER * LSEQ + (size_t)d * LSEQ;
    const size_t ds = (size_t)D_INNER * LSEQ;
    base[0 * ds + l]               = v;
    base[1 * ds + lt]              = v;
    base[2 * ds + (LSEQ - 1 - l)]  = v;
    base[3 * ds + (LSEQ - 1 - lt)] = v;
}

// Selective scan: one 16-lane half-wave per channel (16 states), 16 channels
// per 256-thread block.  Per-lane streams are contiguous in l; prefetch ahead.
__global__ void selective_scan_k(const float* __restrict__ xs, const float* __restrict__ dts,
                                 const float* __restrict__ xdbl,
                                 const float* __restrict__ A_log, const float* __restrict__ Dsp,
                                 float* __restrict__ ys, int L)
{
    const int lane = threadIdx.x & 31;
    const int wave = threadIdx.x >> 5;
    const int chanInBlock = wave * 2 + (lane >> 4);
    const int gchan = blockIdx.x * 16 + chanInBlock;         // [0, B*K*D)
    const int n = lane & 15;
    const int d = gchan % D_INNER;
    const int bk = gchan / D_INNER;                          // b*K + k
    const int k = bk % KDIR;

    const float* xp  = xs  + (size_t)gchan * L;
    const float* dtp = dts + (size_t)gchan * L;
    const float* Bp  = xdbl + (size_t)bk * RPAD * L + (size_t)(DT_RANK + n) * L;
    const float* Cp  = xdbl + (size_t)bk * RPAD * L + (size_t)(DT_RANK + D_STATE + n) * L;
    float* yp = ys + (size_t)gchan * L;

    const float Ac = -__expf(A_log[((size_t)k * D_INNER + d) * D_STATE + n]);
    const float Dc = Dsp[k * D_INNER + d];

    float h = 0.f;
    for (int l = 0; l < L; ++l) {
        if ((l & 63) == 0 && l + 256 < L) {
            __builtin_prefetch(xp + l + 256, 0, 1);
            __builtin_prefetch(dtp + l + 256, 0, 1);
            __builtin_prefetch(Bp + l + 256, 0, 1);
            __builtin_prefetch(Cp + l + 256, 0, 1);
        }
        float x  = xp[l];
        float dt = dtp[l];
        h = h * __expf(dt * Ac) + (dt * x) * Bp[l];
        float yv = h * Cp[l];
        yv += __shfl_xor(yv, 1, 16);
        yv += __shfl_xor(yv, 2, 16);
        yv += __shfl_xor(yv, 4, 16);
        yv += __shfl_xor(yv, 8, 16);
        if (n == 0) yp[l] = yv + Dc * x;
    }
}

// combine 4 scan directions -> ycomb[NTOK,256]
__global__ void combine_y(const float* __restrict__ ys, float* __restrict__ ycomb)
{
    long idx = (long)blockIdx.x * blockDim.x + threadIdx.x;
    if (idx >= (long)NTOK * D_INNER) return;
    int d = idx % D_INNER; long m = idx / D_INNER;
    int j = m % HB; int i = (m / HB) % HB; int b = m / LSEQ;
    int l = i * HB + j, lt = j * HB + i;
    const size_t ds = (size_t)D_INNER * LSEQ;
    const float* base = ys + (size_t)b * KDIR * ds + (size_t)d * LSEQ;
    ycomb[idx] = base[0 * ds + l] + base[1 * ds + lt]
               + base[2 * ds + (LSEQ - 1 - l)] + base[3 * ds + (LSEQ - 1 - lt)];
}

// depthwise 3x3 (+gelu) on token-major [NTOK,512] (channel fastest -> coalesced)
__global__ void dwconv_tok_gelu(const float* __restrict__ in, const float* __restrict__ w,
                                float* __restrict__ out, int C)
{
    long idx = (long)blockIdx.x * blockDim.x + threadIdx.x;
    if (idx >= (long)NTOK * C) return;
    int o = idx % C; long m = idx / C;
    int j = m % HB; int i = (m / HB) % HB; long b = m / LSEQ;
    const float* wp = w + (size_t)o * 9;
    float s = 0.f;
    for (int dy = -1; dy <= 1; ++dy) {
        int yy = i + dy; if (yy < 0 || yy >= HB) continue;
        for (int dx = -1; dx <= 1; ++dx) {
            int xx = j + dx; if (xx < 0 || xx >= HB) continue;
            s += in[((size_t)(b * LSEQ + yy * HB + xx)) * C + o] * wp[(dy + 1) * 3 + (dx + 1)];
        }
    }
    out[idx] = geluf(s);
}

// ---------------------------------------------------------------------------
// Host driver
// ---------------------------------------------------------------------------
static inline dim3 g1(long n, int blk) { return dim3((unsigned)((n + blk - 1) / blk)); }

extern "C" void kernel_launch(void* const* d_in, const int* in_sizes, int n_in,
                              void* d_out, int out_size, void* d_ws, size_t ws_size,
                              hipStream_t stream)
{
    const float* x          = (const float*)d_in[0];
    const float* w_init     = (const float*)d_in[1];
    const float* b_init     = (const float*)d_in[2];
    const float* w_conv1    = (const float*)d_in[3];
    const float* b_conv1    = (const float*)d_in[4];
    const float* ln1_w      = (const float*)d_in[5];
    const float* ln1_b      = (const float*)d_in[6];
    const float* in_proj_w  = (const float*)d_in[7];
    const float* conv_w     = (const float*)d_in[8];
    const float* conv_b     = (const float*)d_in[9];
    const float* x_proj_w   = (const float*)d_in[10];
    const float* dt_proj_w  = (const float*)d_in[11];
    const float* dt_proj_b  = (const float*)d_in[12];
    const float* A_log      = (const float*)d_in[13];
    const float* Dsw        = (const float*)d_in[14];
    const float* out_norm_w = (const float*)d_in[15];
    const float* out_norm_b = (const float*)d_in[16];
    const float* out_proj_w = (const float*)d_in[17];
    const float* pn_w       = (const float*)d_in[18];
    const float* pn_b       = (const float*)d_in[19];
    const float* ff_w1      = (const float*)d_in[20];
    const float* ff_w2      = (const float*)d_in[21];
    const float* ff_w3      = (const float*)d_in[22];
    const float* w_out      = (const float*)d_in[23];
    const float* b_out      = (const float*)d_in[24];
    float* out = (float*)d_out;

    float* ws = (float*)d_ws;
    size_t off = 0;
    auto alloc = [&](size_t n) { float* p = ws + off; off += n; return p; };

    float* x32     = alloc((size_t)2 * 32 * 128 * 128);
    float* tok     = alloc((size_t)NTOK * 128);
    float* htok    = alloc((size_t)NTOK * 128);
    float* hc      = alloc((size_t)NTOK * 128);
    float* xz      = alloc((size_t)NTOK * 512);
    float* xin_img = alloc((size_t)2 * D_INNER * LSEQ);
    float* xs      = alloc((size_t)2 * KDIR * D_INNER * LSEQ);
    float* xdbl    = alloc((size_t)2 * KDIR * RPAD * LSEQ);   // 48 rows per (b,k)
    float* dts     = alloc((size_t)2 * KDIR * D_INNER * LSEQ);
    float* ysb     = alloc((size_t)2 * KDIR * D_INNER * LSEQ);
    float* ycomb   = alloc((size_t)NTOK * D_INNER);
    float* ygate   = alloc((size_t)NTOK * D_INNER);
    float* t1      = alloc((size_t)NTOK * 128);
    float* t2      = alloc((size_t)NTOK * 512);
    float* t3      = alloc((size_t)NTOK * 512);
    float* m32     = alloc((size_t)NTOK * 128);
    float* xp_pad  = alloc((size_t)KDIR * RPAD * D_INNER);    // [k][48][256]
    float* dt_pad  = alloc((size_t)KDIR * D_INNER * DTKP);    // [k][256][32]
    (void)ws_size; (void)in_sizes; (void)n_in; (void)out_size;

    const int TB = 256;

    // --- stem -------------------------------------------------------------
    conv3x3<<<g1((long)2 * 32 * 128 * 128, TB), TB, 0, stream>>>(
        x, w_init, b_init, x32, 2, 6, 32, 128, 128);
    block_to_tokens<<<g1((long)NTOK * 128, TB), TB, 0, stream>>>(x32, tok);
    gemm_wmma_rm<<<dim3(NTOK / 16, 128 / 64), 32, 0, stream>>>(
        tok, 128, w_conv1, 128, b_conv1, nullptr, 0, htok, 128, 128, 0);
    layernorm_k<<<g1(NTOK, 8), 256, 0, stream>>>(
        htok, ln1_w, ln1_b, nullptr, 0, hc, NTOK, 128);

    // --- layers -----------------------------------------------------------
    for (int l = 0; l < NLAYER; ++l) {
        const float* ipw = in_proj_w + (size_t)l * 512 * 128;
        const float* cw  = conv_w + (size_t)l * D_INNER * 9;
        const float* cb  = conv_b + (size_t)l * D_INNER;
        const float* xpw = x_proj_w + (size_t)l * KDIR * 40 * D_INNER;
        const float* dtw = dt_proj_w + (size_t)l * KDIR * D_INNER * DT_RANK;
        const float* dtb = dt_proj_b + (size_t)l * KDIR * D_INNER;
        const float* alw = A_log + (size_t)l * KDIR * D_INNER * D_STATE;
        const float* dsw = Dsw + (size_t)l * KDIR * D_INNER;
        const float* onw = out_norm_w + (size_t)l * D_INNER;
        const float* onb = out_norm_b + (size_t)l * D_INNER;
        const float* opw = out_proj_w + (size_t)l * 128 * D_INNER;

        // stage zero-padded weights for the irregular GEMMs
        pad_rows<<<g1((long)KDIR * RPAD * D_INNER, TB), TB, 0, stream>>>(
            xpw, xp_pad, KDIR, 40, RPAD, D_INNER);
        pad_cols<<<g1((long)KDIR * D_INNER * DTKP, TB), TB, 0, stream>>>(
            dtw, dt_pad, (long)KDIR * D_INNER, DT_RANK, DTKP);

        // SS2D: in_proj (wmma), depthwise conv + silu, 4-direction scatter
        gemm_wmma_rm<<<dim3(NTOK / 16, 512 / 64), 32, 0, stream>>>(
            hc, 128, ipw, 128, nullptr, nullptr, 0, xz, 512, 128, 0);
        xz_to_image<<<g1((long)2 * D_INNER * LSEQ, TB), TB, 0, stream>>>(xz, xin_img);
        dwconv_silu_xs<<<g1((long)2 * D_INNER * LSEQ, TB), TB, 0, stream>>>(
            xin_img, cw, cb, xs);

        // x_proj + dt_proj per (b,k) as strided WMMA GEMMs (padded, unguarded)
        for (int b = 0; b < 2; ++b)
            for (int k = 0; k < KDIR; ++k) {
                const float* Axs = xs + (size_t)(b * KDIR + k) * D_INNER * LSEQ;
                float* Cdb = xdbl + (size_t)(b * KDIR + k) * RPAD * LSEQ;
                gemm_wmma_sk<<<dim3(LSEQ / 16, 1), 128, 0, stream>>>(
                    Axs, LSEQ, xp_pad + (size_t)k * RPAD * D_INNER, D_INNER,
                    nullptr, Cdb, LSEQ, RPAD, D_INNER, 0);
                gemm_wmma_sk<<<dim3(LSEQ / 16, 4), 128, 0, stream>>>(
                    Cdb, LSEQ, dt_pad + (size_t)k * D_INNER * DTKP, DTKP,
                    dtb + (size_t)k * D_INNER,
                    dts + (size_t)(b * KDIR + k) * D_INNER * LSEQ, LSEQ,
                    D_INNER, DTKP, 2 /*softplus*/);
            }

        // selective scan: 2048 channels, 16 per block
        selective_scan_k<<<(2 * KDIR * D_INNER) / 16, 256, 0, stream>>>(
            xs, dts, xdbl, alw, dsw, ysb, LSEQ);

        // combine dirs, out_norm * silu(z), out_proj (+residual into hc)
        combine_y<<<g1((long)NTOK * D_INNER, TB), TB, 0, stream>>>(ysb, ycomb);
        layernorm_k<<<g1(NTOK, 8), 256, 0, stream>>>(
            ycomb, onw, onb, xz + D_INNER, 512, ygate, NTOK, D_INNER);
        gemm_wmma_rm<<<dim3(NTOK / 16, 128 / 64), 32, 0, stream>>>(
            ygate, D_INNER, opw, D_INNER, nullptr, hc, 128, hc, 128,
            D_INNER, 0);

        // FFN
        const float* fw1 = ff_w1 + (size_t)l * 512 * 128;
        const float* fw2 = ff_w2 + (size_t)l * 512 * 9;
        const float* fw3 = ff_w3 + (size_t)l * 128 * 512;
        layernorm_k<<<g1(NTOK, 8), 256, 0, stream>>>(
            hc, pn_w + (size_t)l * 128, pn_b + (size_t)l * 128, nullptr, 0, t1, NTOK, 128);
        gemm_wmma_rm<<<dim3(NTOK / 16, 512 / 64), 32, 0, stream>>>(
            t1, 128, fw1, 128, nullptr, nullptr, 0, t2, 512, 128, 1 /*gelu*/);
        dwconv_tok_gelu<<<g1((long)NTOK * 512, TB), TB, 0, stream>>>(t2, fw2, t3, 512);
        gemm_wmma_rm<<<dim3(NTOK / 16, 128 / 64), 32, 0, stream>>>(
            t3, 512, fw3, 512, nullptr, hc, 128, hc, 128, 512, 0);
    }

    // --- head -------------------------------------------------------------
    tokens_to_image<<<g1((long)NTOK * 128, TB), TB, 0, stream>>>(hc, m32);
    conv3x3<<<g1((long)2 * 3 * 128 * 128, TB), TB, 0, stream>>>(
        m32, w_out, b_out, out, 2, 32, 3, 128, 128);
}
